// MyMultiHeadAttention_34462817583667
// MI455X (gfx1250) — matrix operand order repaired and checked by reading, not compiled, not run on hardware
//
#include <hip/hip_runtime.h>

// ---------------------------------------------------------------------------
// MI455X (gfx1250) fused multi-head attention forward, bf16 WMMA pipeline
// with async-to-LDS double-buffered K/V staging in the attention kernel.
// ---------------------------------------------------------------------------

typedef __bf16 bf16_t;
typedef __attribute__((ext_vector_type(16))) __bf16 v16bf;
typedef __attribute__((ext_vector_type(8)))  float  v8f;
typedef __attribute__((ext_vector_type(4)))  unsigned int u32x4;
// Matches the builtin's parameter pointee type exactly (gcc-style vector).
typedef int v4i_vs __attribute__((vector_size(4 * sizeof(int))));

constexpr int Bn = 2;
constexpr int Sn = 2048;
constexpr int Dn = 1024;
constexpr int Hn = 16;
constexpr int HDn = 64;
constexpr int Mn = Bn * Sn;            // 4096 tokens

// exp(x * 1/sqrt(64)) == exp2(x * 0.125 * log2(e))
#define SCALE_LOG2E 0.18033688011112042f

#ifdef __has_builtin
#if __has_builtin(__builtin_amdgcn_global_load_async_to_lds_b128)
#define HAVE_ASYNC_LDS 1
#endif
#endif

union Frag  { v16bf v; u32x4 u[2]; };
union Pack8 { u32x4 u; bf16_t e[8]; };
union F8    { v8f v; float f[8]; };

__device__ __forceinline__ v8f wmma_bf16(v16bf a, v16bf b, v8f c) {
  // v_wmma_f32_16x16x32_bf16  D = A(16x32) * B(32x16) + C
  return __builtin_amdgcn_wmma_f32_16x16x32_bf16(
      false, a, false, b, (short)0, c, false, false);
}

// A-fragment (16x32 bf16, row-major source, row stride `ld` elements).
// ISA layout: lane holds row (lane&15); K chunks ko..ko+7 / ko+16..ko+23,
// ko = 8*(lane>>4).
__device__ __forceinline__ v16bf load_afrag(const bf16_t* __restrict__ base, int ld) {
  const int lane = threadIdx.x & 31;
  const bf16_t* p = base + (size_t)(lane & 15) * ld + ((lane >> 4) << 3);
  Frag f;
  f.u[0] = *(const u32x4*)(p);
  f.u[1] = *(const u32x4*)(p + 16);
  return f.v;
}

// B-fragment (32x16 bf16, row-major [K x N] source, row stride `ld`).
// ISA layout: lane holds column (lane&15); K rows ko..ko+7 / ko+16..ko+23,
// ko = 8*(lane>>4).
__device__ __forceinline__ v16bf load_bfrag(const bf16_t* __restrict__ base, int ld) {
  const int lane = threadIdx.x & 31;
  const bf16_t* p = base + (size_t)((lane >> 4) << 3) * ld + (lane & 15);
  Frag f;
  #pragma unroll
  for (int s = 0; s < 8; ++s) {
    f.v[s] = p[(size_t)s * ld];
    f.v[s + 8] = p[(size_t)(16 + s) * ld];
  }
  return f.v;
}

// ---------------------------------------------------------------------------
// Elementwise prep kernels
// ---------------------------------------------------------------------------
__global__ void f32_to_bf16_k(const float* __restrict__ src, bf16_t* __restrict__ dst, int n) {
  int i = blockIdx.x * blockDim.x + threadIdx.x;
  int stride = gridDim.x * blockDim.x;
  for (; i < n; i += stride) dst[i] = (bf16_t)src[i];
}

// WT[k*1024 + n] = (bf16) W[n*1024 + k]
__global__ void transpose_to_bf16_k(const float* __restrict__ W, bf16_t* __restrict__ WT) {
  int i = blockIdx.x * blockDim.x + threadIdx.x;
  if (i < Dn * Dn) {
    int kk = i >> 10, nn = i & (Dn - 1);
    WT[i] = (bf16_t)W[(size_t)nn * Dn + kk];
  }
}

__global__ void build_cmask_k(const unsigned char* __restrict__ kpm,
                              const unsigned char* __restrict__ am,
                              unsigned char* __restrict__ cm, int n) {
  int i = blockIdx.x * blockDim.x + threadIdx.x;
  if (i < n) cm[i] = (unsigned char)((kpm[i] | am[i]) ? 1 : 0);
}

// ---------------------------------------------------------------------------
// WMMA GEMM:  C[M=4096, N=1024] = X[M x 1024] * WT[1024 x N] + bias
// Block: 256 threads = 8 waves (4m x 2n), wave tile 32x64 -> block tile
// 128x128; 8 WMMAs per K-step from 6 fragment loads (2x B reuse).
// MODE 0: bf16 out [B,H,S,HD]   (Q)
// MODE 1: bf16 out [B,H,HD,S]   (K,V transposed; contiguous 16B stores)
// MODE 3: f32  out [M,D]        (final projection -> d_out)
// ---------------------------------------------------------------------------
template <int MODE>
__global__ __launch_bounds__(256)
void gemm_bf16_k(const bf16_t* __restrict__ X, const bf16_t* __restrict__ WT,
                 const float* __restrict__ bias, void* __restrict__ outp) {
  const int lane = threadIdx.x & 31;
  const int wave = threadIdx.x >> 5;
  const int wm = wave >> 1, wn = wave & 1;
  const int m0 = blockIdx.x * 128 + wm * 32;
  const int n0 = blockIdx.y * 128 + wn * 64;

  v8f acc[2][4] = {};

  for (int k = 0; k < Dn; k += 32) {
    v16bf a0 = load_afrag(X + (size_t)(m0 +  0) * Dn + k, Dn);
    v16bf a1 = load_afrag(X + (size_t)(m0 + 16) * Dn + k, Dn);
    #pragma unroll
    for (int j = 0; j < 4; ++j) {
      v16bf b = load_bfrag(WT + (size_t)k * Dn + n0 + j * 16, Dn);
      acc[0][j] = wmma_bf16(a0, b, acc[0][j]);
      acc[1][j] = wmma_bf16(a1, b, acc[1][j]);
    }
  }

  const int hl = lane >> 4;
  #pragma unroll
  for (int im = 0; im < 2; ++im) {
    #pragma unroll
    for (int in = 0; in < 4; ++in) {
      const int n = n0 + in * 16 + (lane & 15);
      const float bv = bias[n];
      F8 c; c.v = acc[im][in];
      const int mbase = m0 + im * 16 + hl * 8;   // 8 consecutive tokens (r=0..7)
      if (MODE == 0) {
        bf16_t* O = (bf16_t*)outp;
        const int hh = n >> 6, hd = n & (HDn - 1);
        #pragma unroll
        for (int r = 0; r < 8; ++r) {
          const int m = mbase + r;
          const int bi = m >> 11, s = m & (Sn - 1);
          O[(((size_t)bi * Hn + hh) * Sn + s) * HDn + hd] = (bf16_t)(c.f[r] + bv);
        }
      } else if (MODE == 1) {
        bf16_t* O = (bf16_t*)outp;
        const int hh = n >> 6, hd = n & (HDn - 1);
        const int bi = mbase >> 11, s = mbase & (Sn - 1);
        Pack8 pk;
        #pragma unroll
        for (int r = 0; r < 8; ++r) pk.e[r] = (bf16_t)(c.f[r] + bv);
        *(u32x4*)(O + (((size_t)bi * Hn + hh) * HDn + hd) * Sn + s) = pk.u;
      } else {
        float* O = (float*)outp;
        #pragma unroll
        for (int r = 0; r < 8; ++r)
          O[(size_t)(mbase + r) * Dn + n] = c.f[r] + bv;
      }
    }
  }
}

// ---------------------------------------------------------------------------
// Fused attention: per (b, h, 64-row q block). 4 waves, 16 q-rows per wave.
// 64-key chunks: 8 score WMMAs + mask/exp + 8 PV WMMAs (O^T += V^T P^T).
// K/V chunks staged in LDS (async double-buffer when available); P^T staged
// per-wave in LDS in exact B-fragment layout.
// ---------------------------------------------------------------------------
constexpr int PSTK = 72;   // K/V LDS row stride (64 + pad), 144B: 16B-aligned
constexpr int PSTP = 24;   // P^T LDS row stride (16 + pad),  48B: 16B-aligned

#if HAVE_ASYNC_LDS
__device__ __forceinline__ void async_cp16(const bf16_t* g, bf16_t* l) {
  __builtin_amdgcn_global_load_async_to_lds_b128(
      (__attribute__((address_space(1))) v4i_vs*)(g),
      (__attribute__((address_space(3))) v4i_vs*)(l), 0, 0);
}

__device__ __forceinline__ void stage64(const bf16_t* __restrict__ gK,
                                        const bf16_t* __restrict__ gV,
                                        bf16_t* lK, bf16_t* lV) {
  // Copy K and V chunks [64 d x 64 keys] bf16 -> LDS, 16B per async op.
  #pragma unroll
  for (int i = 0; i < 4; ++i) {
    const int c = threadIdx.x + 128 * i;   // 0..511
    const int d = c >> 3;
    const int hx = (c & 7) * 8;            // element offset within row
    async_cp16(gK + (size_t)d * Sn + hx, lK + d * PSTK + hx);
    async_cp16(gV + (size_t)d * Sn + hx, lV + d * PSTK + hx);
  }
}
#endif

__global__ __launch_bounds__(128)
void attention_k(const bf16_t* __restrict__ Qh, const bf16_t* __restrict__ Kt,
                 const bf16_t* __restrict__ Vt, const unsigned char* __restrict__ cmask,
                 bf16_t* __restrict__ Xo) {
#if HAVE_ASYNC_LDS
  __shared__ bf16_t ldsK[2][64 * PSTK];
  __shared__ bf16_t ldsV[2][64 * PSTK];
#endif
  __shared__ bf16_t pT[4][64 * PSTP];
  __shared__ float  rs[4 * 16];

  const int lane = threadIdx.x & 31;
  const int wave = threadIdx.x >> 5;
  const int hl = lane >> 4;
  const int b = blockIdx.z, h = blockIdx.y;
  const int q0 = blockIdx.x * 64 + wave * 16;

  const size_t bh = (size_t)b * Hn + h;
  const bf16_t* Qb = Qh + (bh * Sn + q0) * HDn;     // [S, HD] slab
  const bf16_t* Kb = Kt + bh * HDn * Sn;            // [HD, S] slab
  const bf16_t* Vb = Vt + bh * HDn * Sn;            // [HD, S] slab
  const unsigned char* cm = cmask + (size_t)b * Sn;
  bf16_t* pt = pT[wave];

  const v16bf qa0 = load_afrag(Qb,      HDn);       // d = 0..31
  const v16bf qa1 = load_afrag(Qb + 32, HDn);       // d = 32..63

  v8f o0 = {}, o1 = {}, o2 = {}, o3 = {};
  float rsum[8] = {0.f, 0.f, 0.f, 0.f, 0.f, 0.f, 0.f, 0.f};

#if HAVE_ASYNC_LDS
  stage64(Kb, Vb, ldsK[0], ldsV[0]);
  asm volatile("s_wait_asynccnt 0" ::: "memory");
  __syncthreads();
  int cur = 0;
#endif

  for (int kk = 0; kk < Sn; kk += 64) {
#if HAVE_ASYNC_LDS
    if (kk + 64 < Sn)
      stage64(Kb + kk + 64, Vb + kk + 64, ldsK[cur ^ 1], ldsV[cur ^ 1]);
    const bf16_t* Ks = ldsK[cur];  const int kld = PSTK;  const int kof = 0;
    const bf16_t* Vs = ldsV[cur];
#else
    const bf16_t* Ks = Kb;         const int kld = Sn;    const int kof = kk;
    const bf16_t* Vs = Vb;
#endif
    // ---- scores + mask + exp + P^T staging -------------------------------
    #pragma unroll
    for (int nt = 0; nt < 4; ++nt) {
      v8f s = {};
      s = wmma_bf16(qa0, load_bfrag(Ks + kof + nt * 16, kld), s);
      s = wmma_bf16(qa1, load_bfrag(Ks + (size_t)32 * kld + kof + nt * 16, kld), s);
      const int col = kk + nt * 16 + (lane & 15);
      const bool masked = (cm[col] != 0);            // -inf score -> exp = 0
      F8 sv; sv.v = s;
      Pack8 pk;
      #pragma unroll
      for (int r = 0; r < 8; ++r) {
        float e = masked ? 0.0f : exp2f(sv.f[r] * SCALE_LOG2E);
        rsum[r] += e;
        pk.e[r] = (bf16_t)e;
      }
      // C layout: lane holds key column nt*16+(lane&15), rows q = hl*8 + r.
      *(u32x4*)(pt + (size_t)(nt * 16 + (lane & 15)) * PSTP + hl * 8) = pk.u;
    }
    __syncthreads();
    // ---- O^T += V^T * P^T ------------------------------------------------
    #pragma unroll
    for (int ks = 0; ks < 2; ++ks) {
      const v16bf pb = load_bfrag(pt + (size_t)(ks * 32) * PSTP, PSTP);
      const int ko = kof + ks * 32;
      o0 = wmma_bf16(load_afrag(Vs + (size_t) 0 * kld + ko, kld), pb, o0);
      o1 = wmma_bf16(load_afrag(Vs + (size_t)16 * kld + ko, kld), pb, o1);
      o2 = wmma_bf16(load_afrag(Vs + (size_t)32 * kld + ko, kld), pb, o2);
      o3 = wmma_bf16(load_afrag(Vs + (size_t)48 * kld + ko, kld), pb, o3);
    }
#if HAVE_ASYNC_LDS
    asm volatile("s_wait_asynccnt 0" ::: "memory");
    cur ^= 1;
#endif
    __syncthreads();
  }

  // Row sums: reduce per-lane partials across the 16 lanes of each half.
  #pragma unroll
  for (int r = 0; r < 8; ++r) {
    #pragma unroll
    for (int ofs = 1; ofs < 16; ofs <<= 1)
      rsum[r] += __shfl_xor(rsum[r], ofs, 32);
  }
  if ((lane & 15) == 0) {
    #pragma unroll
    for (int r = 0; r < 8; ++r) rs[wave * 16 + hl * 8 + r] = rsum[r];
  }
  __syncthreads();
  const int q = lane & 15;
  const float inv = 1.0f / rs[wave * 16 + q];       // O^T column q = lane index

  // O^T C-layout: lane holds (d = dt*16 + hl*8 + r, q); d contiguous in r
  // -> 16B stores into x[b, s=q0+q, h*64 + d].
  bf16_t* orow = Xo + ((size_t)b * Sn + q0 + q) * Dn + (size_t)h * HDn;
  v8f os[4] = {o0, o1, o2, o3};
  #pragma unroll
  for (int dt = 0; dt < 4; ++dt) {
    F8 ov; ov.v = os[dt];
    Pack8 pk;
    #pragma unroll
    for (int r = 0; r < 8; ++r) pk.e[r] = (bf16_t)(ov.f[r] * inv);
    *(u32x4*)(orow + dt * 16 + hl * 8) = pk.u;
  }
}

// ---------------------------------------------------------------------------
extern "C" void kernel_launch(void* const* d_in, const int* in_sizes, int n_in,
                              void* d_out, int out_size, void* d_ws, size_t ws_size,
                              hipStream_t stream) {
  (void)in_sizes; (void)n_in; (void)out_size; (void)ws_size;

  const float* q  = (const float*)d_in[0];
  const float* k  = (const float*)d_in[1];
  const float* v  = (const float*)d_in[2];
  const unsigned char* kpm = (const unsigned char*)d_in[3];  // jax bool = 1 byte
  const unsigned char* am  = (const unsigned char*)d_in[4];
  const float* Wq = (const float*)d_in[6];
  const float* bq = (const float*)d_in[7];
  const float* Wk = (const float*)d_in[8];
  const float* bk = (const float*)d_in[9];
  const float* Wv = (const float*)d_in[10];
  const float* bv = (const float*)d_in[11];
  const float* Wo = (const float*)d_in[12];
  const float* bo = (const float*)d_in[13];

  char* ws = (char*)d_ws;
  size_t off = 0;
  auto wsalloc = [&](size_t bytes) -> char* {
    char* p = ws + off;
    off += (bytes + 255) & ~(size_t)255;
    return p;
  };
  const size_t actB = (size_t)Mn * Dn * sizeof(bf16_t);   // 8 MB
  const size_t wB   = (size_t)Dn * Dn * sizeof(bf16_t);   // 2 MB
  bf16_t* qb  = (bf16_t*)wsalloc(actB);
  bf16_t* kb  = (bf16_t*)wsalloc(actB);
  bf16_t* vb  = (bf16_t*)wsalloc(actB);
  bf16_t* WqT = (bf16_t*)wsalloc(wB);
  bf16_t* WkT = (bf16_t*)wsalloc(wB);
  bf16_t* WvT = (bf16_t*)wsalloc(wB);
  bf16_t* WoT = (bf16_t*)wsalloc(wB);
  bf16_t* Qh  = (bf16_t*)wsalloc(actB);   // [B,H,S,HD]
  bf16_t* Ktb = (bf16_t*)wsalloc(actB);   // [B,H,HD,S]
  bf16_t* Vtb = (bf16_t*)wsalloc(actB);   // [B,H,HD,S]
  bf16_t* Xab = (bf16_t*)wsalloc(actB);   // attention output [B,S,D]
  unsigned char* cmb = (unsigned char*)wsalloc((size_t)Bn * Sn);

  const int nAct = Mn * Dn;
  f32_to_bf16_k<<<(nAct + 255) / 256, 256, 0, stream>>>(q, qb, nAct);
  f32_to_bf16_k<<<(nAct + 255) / 256, 256, 0, stream>>>(k, kb, nAct);
  f32_to_bf16_k<<<(nAct + 255) / 256, 256, 0, stream>>>(v, vb, nAct);

  const int nW = Dn * Dn;
  transpose_to_bf16_k<<<(nW + 255) / 256, 256, 0, stream>>>(Wq, WqT);
  transpose_to_bf16_k<<<(nW + 255) / 256, 256, 0, stream>>>(Wk, WkT);
  transpose_to_bf16_k<<<(nW + 255) / 256, 256, 0, stream>>>(Wv, WvT);
  transpose_to_bf16_k<<<(nW + 255) / 256, 256, 0, stream>>>(Wo, WoT);

  build_cmask_k<<<(Bn * Sn + 255) / 256, 256, 0, stream>>>(kpm, am, cmb, Bn * Sn);

  dim3 gg(Mn / 128, Dn / 128);
  gemm_bf16_k<0><<<gg, 256, 0, stream>>>(qb, WqT, bq, Qh);
  gemm_bf16_k<1><<<gg, 256, 0, stream>>>(kb, WkT, bk, Ktb);
  gemm_bf16_k<1><<<gg, 256, 0, stream>>>(vb, WvT, bv, Vtb);

  attention_k<<<dim3(Sn / 64, Hn, Bn), 128, 0, stream>>>(Qh, Ktb, Vtb, cmb, Xab);

  gemm_bf16_k<3><<<gg, 256, 0, stream>>>(Xab, WoT, bo, d_out);
}
